// SelfAttention_10917806866648
// MI455X (gfx1250) — matrix operand order repaired and checked by reading, not compiled, not run on hardware
//
#include <hip/hip_runtime.h>
#include <hip/hip_bf16.h>

// Problem constants
#define BB  2
#define SS  2048
#define DD  512
#define HH  8
#define DKK 64
#define BSR (BB*SS)   // 4096 token rows

typedef __bf16 bf16_t;
typedef __bf16 bf16x16 __attribute__((ext_vector_type(16)));
typedef __bf16 bf16x8  __attribute__((ext_vector_type(8)));
typedef __bf16 bf16x4  __attribute__((ext_vector_type(4)));
typedef float  f32x8   __attribute__((ext_vector_type(8)));

// --------------------------------------------------------------------------
// gfx1250 async global->LDS copy via inline asm (ISA 10. / 15.18.3).
//   global_load_async_to_lds_b128 vdst, vaddr, off
// vdst = VGPR with wave-relative LDS byte address (low 32 bits of the generic
// pointer, per aperture rules), vaddr = 64-bit global address.
// Tracked by ASYNCcnt; async loads complete in order, so waiting
// ASYNCcnt <= 4 after issuing 4 new copies proves the previous 4 landed.
// --------------------------------------------------------------------------
#if defined(__gfx1250__)
#define HAVE_ASYNC_LDS 1
#endif

__device__ __forceinline__ void copy16_to_lds(bf16_t* lds_dst, const bf16_t* gsrc) {
#if defined(HAVE_ASYNC_LDS)
    unsigned lds32 = (unsigned)(size_t)lds_dst;               // LDS byte offset
    unsigned long long ga = (unsigned long long)(size_t)gsrc; // global address
    asm volatile("global_load_async_to_lds_b128 %0, %1, off"
                 :: "v"(lds32), "v"(ga)
                 : "memory");
#else
    *(bf16x8*)lds_dst = *(const bf16x8*)gsrc;
#endif
}

__device__ __forceinline__ void wait_async_le4() {   // previous stage done
#if defined(HAVE_ASYNC_LDS)
    asm volatile("s_wait_asynccnt 0x4" ::: "memory");
#endif
}

__device__ __forceinline__ void wait_async_le0() {   // everything done
#if defined(HAVE_ASYNC_LDS)
    asm volatile("s_wait_asynccnt 0x0" ::: "memory");
#endif
}

// ---------------------------------------------------------------------------
// WMMA operand loaders (CDNA5 16x16x32 bf16 layouts, wave32)
//
// A (16x32, MxK): lane = m + 16*g holds row m:
//   elements[0..7]  = A[m][ 8*g + 0..7]
//   elements[8..15] = A[m][16 + 8*g + 0..7]
// B (32x16, KxN): lane = n + 16*g holds col n:
//   elements[0..7] = B[8*g + 0..7][n], elements[8..15] = B[16 + 8*g + 0..7][n]
// C/D (16x16 f32): lane = n + 16*g, element r = (row r + 8*g, col n)
// ---------------------------------------------------------------------------
__device__ __forceinline__ bf16x16 load_a16(const bf16_t* base, int stride) {
    const int lane = threadIdx.x & 31;
    const int m = lane & 15, g = lane >> 4;
    const bf16_t* p0 = base + m * stride + g * 8;
    bf16x8 lo = *(const bf16x8*)p0;
    bf16x8 hi = *(const bf16x8*)(p0 + 16);
    bf16x16 r;
#pragma unroll
    for (int i = 0; i < 8; ++i) { r[i] = lo[i]; r[8 + i] = hi[i]; }
    return r;
}

__device__ __forceinline__ bf16x16 load_b16(const bf16_t* base, int stride) {
    const int lane = threadIdx.x & 31;
    const int n = lane & 15, g = lane >> 4;
    const bf16_t* p0 = base + n * stride + g * 8;
    bf16x8 lo = *(const bf16x8*)p0;
    bf16x8 hi = *(const bf16x8*)(p0 + 16);
    bf16x16 r;
#pragma unroll
    for (int i = 0; i < 8; ++i) { r[i] = lo[i]; r[8 + i] = hi[i]; }
    return r;
}

__device__ __forceinline__ f32x8 wmma_bf16(bf16x16 a, bf16x16 b, f32x8 c) {
    return __builtin_amdgcn_wmma_f32_16x16x32_bf16(
        /*neg_a=*/false, a, /*neg_b=*/false, b,
        /*c_mod=*/(short)0, c, /*reuse_a=*/false, /*reuse_b=*/false);
}

// ---------------------------------------------------------------------------
// Kernel 1: f32 -> bf16 conversion (vectorized x4)
// ---------------------------------------------------------------------------
__global__ void __launch_bounds__(256) cvt_f32_bf16(const float* __restrict__ s,
                                                    bf16_t* __restrict__ d, int n4) {
    int i = blockIdx.x * blockDim.x + threadIdx.x;
    if (i < n4) {
        float4 v = ((const float4*)s)[i];
        bf16x4 o;
        o[0] = (bf16_t)v.x; o[1] = (bf16_t)v.y;
        o[2] = (bf16_t)v.z; o[3] = (bf16_t)v.w;
        ((bf16x4*)d)[i] = o;
    }
}

// ---------------------------------------------------------------------------
// Head-major store of one 16x16 f32 C tile (bf16 result).
// which: 0 -> qh [B,H,S,DK], 1 -> kh [B,H,S,DK], 2 -> vT [B,H,DK,S]
// ---------------------------------------------------------------------------
__device__ __forceinline__ void store_head_tile(int which, f32x8 c, int m0, int n0,
                                                bf16_t* __restrict__ qh,
                                                bf16_t* __restrict__ kh,
                                                bf16_t* __restrict__ vT) {
    const int lane = threadIdx.x & 31;
    const int n = lane & 15, g = lane >> 4;
    const int col = n0 + n;                 // 0..511
    const int h = col >> 6, dk = col & 63;
#pragma unroll
    for (int r = 0; r < 8; ++r) {
        int row = m0 + r + 8 * g;           // 0..4095
        int bb = row >> 11, s = row & (SS - 1);
        bf16_t v = (bf16_t)c[r];
        if (which == 2) {
            vT[(((size_t)bb * HH + h) * DKK + dk) * SS + s] = v;
        } else {
            bf16_t* dst = (which == 0) ? qh : kh;
            dst[(((size_t)bb * HH + h) * SS + s) * DKK + dk] = v;
        }
    }
}

// ---------------------------------------------------------------------------
// Kernel 2: QKV projection  out = x @ W.T
// Register-blocked: each wave computes a 64x32 output tile
// (4 A-tiles x 2 B-tiles -> 8 WMMAs per 32-wide K step).
// blockIdx.z: 0=Q, 1=K, 2=V.  Q,K -> [B,H,S,DK]; V -> transposed [B,H,DK,S].
// ---------------------------------------------------------------------------
__global__ void __launch_bounds__(128) qkv_proj_kernel(
    const bf16_t* __restrict__ qx, const bf16_t* __restrict__ kx, const bf16_t* __restrict__ vx,
    const bf16_t* __restrict__ wq, const bf16_t* __restrict__ wk, const bf16_t* __restrict__ wv,
    bf16_t* __restrict__ qh, bf16_t* __restrict__ kh, bf16_t* __restrict__ vT)
{
    const int which = blockIdx.z;
    const bf16_t* x = (which == 0) ? qx : (which == 1) ? kx : vx;
    const bf16_t* w = (which == 0) ? wq : (which == 1) ? wk : wv;

    const int wave = threadIdx.x >> 5;
    const int tile = blockIdx.x * 4 + wave;     // 1024 tiles
    const int NT = DD / 32;                     // 16 col tiles
    const int m0 = (tile / NT) * 64;
    const int n0 = (tile % NT) * 32;

    f32x8 c[4][2] = {};
#pragma unroll 2
    for (int k0 = 0; k0 < DD; k0 += 32) {
        bf16x16 a0 = load_a16(x + (size_t)(m0 +  0) * DD + k0, DD);
        bf16x16 a1 = load_a16(x + (size_t)(m0 + 16) * DD + k0, DD);
        bf16x16 a2 = load_a16(x + (size_t)(m0 + 32) * DD + k0, DD);
        bf16x16 a3 = load_a16(x + (size_t)(m0 + 48) * DD + k0, DD);
        bf16x16 b0 = load_b16(w + (size_t)(n0 +  0) * DD + k0, DD);
        bf16x16 b1 = load_b16(w + (size_t)(n0 + 16) * DD + k0, DD);
        c[0][0] = wmma_bf16(a0, b0, c[0][0]);  c[0][1] = wmma_bf16(a0, b1, c[0][1]);
        c[1][0] = wmma_bf16(a1, b0, c[1][0]);  c[1][1] = wmma_bf16(a1, b1, c[1][1]);
        c[2][0] = wmma_bf16(a2, b0, c[2][0]);  c[2][1] = wmma_bf16(a2, b1, c[2][1]);
        c[3][0] = wmma_bf16(a3, b0, c[3][0]);  c[3][1] = wmma_bf16(a3, b1, c[3][1]);
    }

#pragma unroll
    for (int i = 0; i < 4; ++i)
#pragma unroll
        for (int j = 0; j < 2; ++j)
            store_head_tile(which, c[i][j], m0 + 16 * i, n0 + 16 * j, qh, kh, vT);
}

// ---------------------------------------------------------------------------
// Kernel 3: causal flash attention, double-buffered async LDS staging.
// One block (4 waves) owns 64 consecutive q-rows of one (b,h); wave w takes
// rows [Q0+16w, Q0+16w+16). Per 32-key chunk the block async-copies
// K (32x64) and V^T (64x32) into the alternate LDS buffer while computing on
// the current one; all waves consume via ds_load.
// ---------------------------------------------------------------------------
__global__ void __launch_bounds__(128) attn_kernel(
    const bf16_t* __restrict__ qh, const bf16_t* __restrict__ kh,
    const bf16_t* __restrict__ vT, bf16_t* __restrict__ ybf)
{
    __shared__ bf16_t Kt[2][32 * 64];       // [buf][key][dk]   8 KB
    __shared__ bf16_t Vt[2][64 * 32];       // [buf][dk][key]   8 KB
    __shared__ bf16_t pbuf[4][16 * 32];     // per-wave P       4 KB

    const int tid  = threadIdx.x;
    const int wave = tid >> 5;
    const int lane = tid & 31;
    const int n = lane & 15, g = lane >> 4;

    const int bh = blockIdx.x >> 5;         // 0..15
    const int Q0 = (blockIdx.x & 31) * 64;  // block q-base
    const int q0 = Q0 + wave * 16;          // wave q-base

    const bf16_t* qbase = qh + (size_t)bh * SS * DKK + (size_t)q0 * DKK;
    const bf16_t* kbase = kh + (size_t)bh * SS * DKK;
    const bf16_t* vbase = vT + (size_t)bh * DKK * SS;

    // Q tile as two A operands (dk 0..31 and 32..63)
    bf16x16 aq0 = load_a16(qbase + 0, DKK);
    bf16x16 aq1 = load_a16(qbase + 32, DKK);

    f32x8 acc0 = {}, acc1 = {}, acc2 = {}, acc3 = {};
    float mrow[8], lrow[8];
#pragma unroll
    for (int r = 0; r < 8; ++r) { mrow[r] = -__builtin_inff(); lrow[r] = 0.0f; }

    bf16_t* pb = pbuf[wave];

    // staging coordinates (per thread, 16B granules)
    const int kr = tid >> 3, kc = (tid & 7) * 8;   // K rows: 128B each
    const int vr = tid >> 2, vc = (tid & 3) * 8;   // V rows: 64B each

    // issue the 4 async b128 copies for one 32-key chunk into buffer `buf`
    auto stage = [&](int k0, int buf) {
        copy16_to_lds(&Kt[buf][(kr +  0) * 64 + kc], kbase + (size_t)(k0 + kr +  0) * DKK + kc);
        copy16_to_lds(&Kt[buf][(kr + 16) * 64 + kc], kbase + (size_t)(k0 + kr + 16) * DKK + kc);
        copy16_to_lds(&Vt[buf][(vr +  0) * 32 + vc], vbase + (size_t)(vr +  0) * SS + k0 + vc);
        copy16_to_lds(&Vt[buf][(vr + 32) * 32 + vc], vbase + (size_t)(vr + 32) * SS + k0 + vc);
    };

    const int nch = (Q0 + 64) / 32;         // block-uniform chunk count (>= 2)
    stage(0, 0);                            // prologue

    for (int i = 0; i < nch; ++i) {
        const int k0 = i * 32;
        const int buf = i & 1;

        if (i + 1 < nch) {                  // pipeline next chunk into other buffer
            stage((i + 1) * 32, buf ^ 1);
            wait_async_le4();               // current chunk's 4 copies complete
        } else {
            wait_async_le0();
        }
        __syncthreads();

        if (k0 < q0 + 16) {                 // wave-uniform causal skip
            const bf16_t* Kc = Kt[buf];
            const bf16_t* Vc = Vt[buf];

            // ---- scores: two 16x16 tiles over keys [k0, k0+32) from LDS ----
            f32x8 s0 = {}, s1 = {};
            s0 = wmma_bf16(aq0, load_b16(Kc +  0 * 64 +  0, 64), s0);
            s0 = wmma_bf16(aq1, load_b16(Kc +  0 * 64 + 32, 64), s0);
            s1 = wmma_bf16(aq0, load_b16(Kc + 16 * 64 +  0, 64), s1);
            s1 = wmma_bf16(aq1, load_b16(Kc + 16 * 64 + 32, 64), s1);

            // ---- scale, causal mask, online softmax ----
#pragma unroll
            for (int r = 0; r < 8; ++r) {
                const int rowg = q0 + r + 8 * g;
                float x0 = s0[r] * 0.125f;  // 1/sqrt(64)
                float x1 = s1[r] * 0.125f;
                if (k0 + n      > rowg) x0 = -1e10f;
                if (k0 + 16 + n > rowg) x1 = -1e10f;

                float mx = fmaxf(x0, x1);
#pragma unroll
                for (int off = 1; off < 16; off <<= 1)
                    mx = fmaxf(mx, __shfl_xor(mx, off, 32));
                float mn = fmaxf(mrow[r], mx);
                float p0 = __expf(x0 - mn);
                float p1 = __expf(x1 - mn);
                float corr = __expf(mrow[r] - mn);
                float rs = p0 + p1;
#pragma unroll
                for (int off = 1; off < 16; off <<= 1)
                    rs += __shfl_xor(rs, off, 32);
                lrow[r] = lrow[r] * corr + rs;
                mrow[r] = mn;
                acc0[r] *= corr; acc1[r] *= corr; acc2[r] *= corr; acc3[r] *= corr;

                // C-layout -> row-major P tile in LDS (bf16)
                pb[(r + 8 * g) * 32 + n]      = (bf16_t)p0;
                pb[(r + 8 * g) * 32 + 16 + n] = (bf16_t)p1;
            }

            // ---- PV: P(16x32) x V(32x64), V^T tiles from LDS ----
            bf16x16 ap = load_a16(pb, 32);
            acc0 = wmma_bf16(ap, load_b16(Vc +  0 * 32, 32), acc0);
            acc1 = wmma_bf16(ap, load_b16(Vc + 16 * 32, 32), acc1);
            acc2 = wmma_bf16(ap, load_b16(Vc + 32 * 32, 32), acc2);
            acc3 = wmma_bf16(ap, load_b16(Vc + 48 * 32, 32), acc3);
        }
        __syncthreads();                    // all reads of buf done before reuse
    }

    // ---- normalize and store y (bf16, model-major [B,S,D]) ----
    const int bb = bh / HH, h = bh % HH;
#pragma unroll
    for (int r = 0; r < 8; ++r) {
        int s = q0 + r + 8 * g;
        float inv = 1.0f / lrow[r];
        size_t rowoff = ((size_t)bb * SS + s) * DD + (size_t)h * DKK;
        ybf[rowoff +  0 + n] = (bf16_t)(acc0[r] * inv);
        ybf[rowoff + 16 + n] = (bf16_t)(acc1[r] * inv);
        ybf[rowoff + 32 + n] = (bf16_t)(acc2[r] * inv);
        ybf[rowoff + 48 + n] = (bf16_t)(acc3[r] * inv);
    }
}

// ---------------------------------------------------------------------------
// Kernel 4: output projection  out = y @ Wp.T + bp  (f32 output)
// Register-blocked 64x32 per wave, same as kernel 2.
// ---------------------------------------------------------------------------
__global__ void __launch_bounds__(128) out_proj_kernel(
    const bf16_t* __restrict__ y, const bf16_t* __restrict__ wp,
    const float* __restrict__ bp, float* __restrict__ out)
{
    const int wave = threadIdx.x >> 5;
    const int tile = blockIdx.x * 4 + wave;
    const int NT = DD / 32;
    const int m0 = (tile / NT) * 64;
    const int n0 = (tile % NT) * 32;

    f32x8 c[4][2] = {};
#pragma unroll 2
    for (int k0 = 0; k0 < DD; k0 += 32) {
        bf16x16 a0 = load_a16(y + (size_t)(m0 +  0) * DD + k0, DD);
        bf16x16 a1 = load_a16(y + (size_t)(m0 + 16) * DD + k0, DD);
        bf16x16 a2 = load_a16(y + (size_t)(m0 + 32) * DD + k0, DD);
        bf16x16 a3 = load_a16(y + (size_t)(m0 + 48) * DD + k0, DD);
        bf16x16 b0 = load_b16(wp + (size_t)(n0 +  0) * DD + k0, DD);
        bf16x16 b1 = load_b16(wp + (size_t)(n0 + 16) * DD + k0, DD);
        c[0][0] = wmma_bf16(a0, b0, c[0][0]);  c[0][1] = wmma_bf16(a0, b1, c[0][1]);
        c[1][0] = wmma_bf16(a1, b0, c[1][0]);  c[1][1] = wmma_bf16(a1, b1, c[1][1]);
        c[2][0] = wmma_bf16(a2, b0, c[2][0]);  c[2][1] = wmma_bf16(a2, b1, c[2][1]);
        c[3][0] = wmma_bf16(a3, b0, c[3][0]);  c[3][1] = wmma_bf16(a3, b1, c[3][1]);
    }

    const int lane = threadIdx.x & 31;
    const int n = lane & 15, g = lane >> 4;
#pragma unroll
    for (int i = 0; i < 4; ++i) {
#pragma unroll
        for (int j = 0; j < 2; ++j) {
            float bias = bp[n0 + 16 * j + n];
#pragma unroll
            for (int r = 0; r < 8; ++r)
                out[(size_t)(m0 + 16 * i + r + 8 * g) * DD + n0 + 16 * j + n] =
                    c[i][j][r] + bias;
        }
    }
}

// ---------------------------------------------------------------------------
// Host launcher
// ---------------------------------------------------------------------------
extern "C" void kernel_launch(void* const* d_in, const int* in_sizes, int n_in,
                              void* d_out, int out_size, void* d_ws, size_t ws_size,
                              hipStream_t stream) {
    const float* q  = (const float*)d_in[0];
    const float* k  = (const float*)d_in[1];
    const float* v  = (const float*)d_in[2];
    const float* Wq = (const float*)d_in[3];
    const float* Wk = (const float*)d_in[4];
    const float* Wv = (const float*)d_in[5];
    const float* Wp = (const float*)d_in[6];
    const float* bp = (const float*)d_in[7];
    float* out = (float*)d_out;

    const int NX = BSR * DD;      // 2,097,152 elements per activation
    const int NW = DD * DD;       //   262,144 elements per weight

    char* ws = (char*)d_ws;
    size_t off = 0;
    auto take = [&](size_t bytes) {
        void* p = ws + off;
        off += (bytes + 255) & ~(size_t)255;
        return p;
    };
    bf16_t* qxb = (bf16_t*)take((size_t)NX * 2);
    bf16_t* kxb = (bf16_t*)take((size_t)NX * 2);
    bf16_t* vxb = (bf16_t*)take((size_t)NX * 2);
    bf16_t* wqb = (bf16_t*)take((size_t)NW * 2);
    bf16_t* wkb = (bf16_t*)take((size_t)NW * 2);
    bf16_t* wvb = (bf16_t*)take((size_t)NW * 2);
    bf16_t* wpb = (bf16_t*)take((size_t)NW * 2);
    bf16_t* qhb = (bf16_t*)take((size_t)NX * 2);  // [B,H,S,DK]
    bf16_t* khb = (bf16_t*)take((size_t)NX * 2);  // [B,H,S,DK]
    bf16_t* vTb = (bf16_t*)take((size_t)NX * 2);  // [B,H,DK,S]
    bf16_t* ybf = (bf16_t*)take((size_t)NX * 2);  // [B,S,D]

    // 1) fp32 -> bf16
    cvt_f32_bf16<<<NX / 4 / 256, 256, 0, stream>>>(q,  qxb, NX / 4);
    cvt_f32_bf16<<<NX / 4 / 256, 256, 0, stream>>>(k,  kxb, NX / 4);
    cvt_f32_bf16<<<NX / 4 / 256, 256, 0, stream>>>(v,  vxb, NX / 4);
    cvt_f32_bf16<<<NW / 4 / 256, 256, 0, stream>>>(Wq, wqb, NW / 4);
    cvt_f32_bf16<<<NW / 4 / 256, 256, 0, stream>>>(Wk, wkb, NW / 4);
    cvt_f32_bf16<<<NW / 4 / 256, 256, 0, stream>>>(Wv, wvb, NW / 4);
    cvt_f32_bf16<<<NW / 4 / 256, 256, 0, stream>>>(Wp, wpb, NW / 4);

    // 2) QKV projections: 1024 64x32 tiles / 4 waves per block, z = {Q,K,V}
    qkv_proj_kernel<<<dim3((BSR / 64) * (DD / 32) / 4, 1, 3), 128, 0, stream>>>(
        qxb, kxb, vxb, wqb, wkb, wvb, qhb, khb, vTb);

    // 3) attention: one block per (b,h,64 q-rows)
    attn_kernel<<<BB * HH * (SS / 64), 128, 0, stream>>>(qhb, khb, vTb, ybf);

    // 4) output projection + bias
    out_proj_kernel<<<(BSR / 64) * (DD / 32) / 4, 128, 0, stream>>>(ybf, wpb, bp, out);
}
